// DeeperDynamicEdgeNet_6287832121545
// MI455X (gfx1250) — compile-verified, matrix operand order checked by reading, not run on hardware
//
#include <hip/hip_runtime.h>
#include <hip/hip_bf16.h>

typedef __attribute__((ext_vector_type(16))) _Float16 v16h;
typedef __attribute__((ext_vector_type(8)))  float    v8f;

#define BN_EPS 1e-5f

// =====================================================================
// Small generic kernels
// =====================================================================
__global__ void zero_f(float* p, int n) {
  int i = blockIdx.x * blockDim.x + threadIdx.x;
  if (i < n) p[i] = 0.f;
}

// column stats for C<=4 channels (bn over x / u)
__global__ void colstats(const float* __restrict__ X, int M, int C,
                         float* S1, float* S2) {
  __shared__ float bs[4], bq[4];
  if (threadIdx.x < 4) { bs[threadIdx.x] = 0.f; bq[threadIdx.x] = 0.f; }
  __syncthreads();
  float s[4] = {0,0,0,0}, q[4] = {0,0,0,0};
  for (int m = blockIdx.x * blockDim.x + threadIdx.x; m < M;
       m += gridDim.x * blockDim.x) {
#pragma unroll
    for (int c = 0; c < 4; c++)
      if (c < C) { float v = X[m * C + c]; s[c] += v; q[c] += v * v; }
  }
#pragma unroll
  for (int c = 0; c < 4; c++)
    if (c < C) { atomicAdd(&bs[c], s[c]); atomicAdd(&bq[c], q[c]); }
  __syncthreads();
  if (threadIdx.x < C) {
    atomicAdd(&S1[threadIdx.x], bs[threadIdx.x]);
    atomicAdd(&S2[threadIdx.x], bq[threadIdx.x]);
  }
}

__global__ void colnorm(const float* __restrict__ X, int M, int C,
                        const float* __restrict__ S1, const float* __restrict__ S2,
                        const float* __restrict__ g, const float* __restrict__ b,
                        float* __restrict__ Y) {
  int i = blockIdx.x * blockDim.x + threadIdx.x;
  if (i >= M * C) return;
  int c = i % C;
  float mu = S1[c] / (float)M;
  float var = S2[c] / (float)M - mu * mu;
  Y[i] = g[c] * (X[i] - mu) * rsqrtf(var + BN_EPS) + b[c];
}

// BN stats over M rows for channel blockIdx.x; outputs folded a/shift
__global__ void bn_stats_rows(const float* __restrict__ X, int M, int N,
                              const float* __restrict__ g, const float* __restrict__ be,
                              float* __restrict__ a, float* __restrict__ sh) {
  int n = blockIdx.x;
  __shared__ float s1[256], s2[256];
  float s = 0.f, q = 0.f;
  for (int m = threadIdx.x; m < M; m += blockDim.x) {
    float v = X[m * N + n]; s += v; q += v * v;
  }
  s1[threadIdx.x] = s; s2[threadIdx.x] = q;
  __syncthreads();
  for (int o = 128; o > 0; o >>= 1) {
    if (threadIdx.x < o) { s1[threadIdx.x] += s1[threadIdx.x + o];
                           s2[threadIdx.x] += s2[threadIdx.x + o]; }
    __syncthreads();
  }
  if (threadIdx.x == 0) {
    float mu = s1[0] / (float)M;
    float var = s2[0] / (float)M - mu * mu;
    float av = g[n] * rsqrtf(var + BN_EPS);
    a[n] = av; sh[n] = be[n] - av * mu;
  }
}

__global__ void bn_relu(float* X, int total, int N,
                        const float* __restrict__ a, const float* __restrict__ sh) {
  int i = blockIdx.x * blockDim.x + threadIdx.x;
  if (i >= total) return;
  int n = i % N;
  X[i] = fmaxf(0.f, a[n] * X[i] + sh[n]);
}

__global__ void gemm_bias(const float* __restrict__ A, const float* __restrict__ W,
                          const float* __restrict__ b, float* __restrict__ C,
                          int M, int Kd, int N) {
  int id = blockIdx.x * blockDim.x + threadIdx.x;
  if (id >= M * N) return;
  int m = id / N, n = id % N;
  float s = b[n];
  for (int k = 0; k < Kd; k++) s += A[m * Kd + k] * W[k * N + n];
  C[id] = s;
}

__global__ void final_out(const float* __restrict__ X /*512x256*/,
                          const float* __restrict__ W3 /*256*/,
                          const float* __restrict__ b3, float* __restrict__ out) {
  int g = blockIdx.x;
  __shared__ float red[256];
  red[threadIdx.x] = X[g * 256 + threadIdx.x] * W3[threadIdx.x];
  __syncthreads();
  for (int o = 128; o > 0; o >>= 1) {
    if (threadIdx.x < o) red[threadIdx.x] += red[threadIdx.x + o];
    __syncthreads();
  }
  if (threadIdx.x == 0) out[g] = red[0] + b3[0];
}

__global__ void conv_stats_fin(const float* __restrict__ S1, const float* __restrict__ S2,
                               const float* __restrict__ g, const float* __restrict__ be,
                               float* __restrict__ a, float* __restrict__ sh,
                               int DH, float invN) {
  int c = blockIdx.x * blockDim.x + threadIdx.x;
  if (c < DH) {
    float mu = S1[c] * invN;
    float var = S2[c] * invN - mu * mu;   // linear bias cancels in BN folding
    float av = g[c] * rsqrtf(var + BN_EPS);
    a[c] = av; sh[c] = be[c] - av * mu;
  }
}

// =====================================================================
// kNN: one block (128 threads) per graph, top-16 smallest d2 (self incl.)
// =====================================================================
__global__ __launch_bounds__(128)
void knn_kernel(const float* __restrict__ F, int DC, int* __restrict__ IDX) {
  __shared__ float LF[128][68];
  int g = blockIdx.x, tid = threadIdx.x;
  for (int t = tid; t < 128 * DC; t += 128) {
    int r = t / DC;
    LF[r][t - r * DC] = F[(size_t)g * 128 * DC + t];
  }
  __syncthreads();
  float bd[16]; int bi[16];
#pragma unroll
  for (int t = 0; t < 16; t++) { bd[t] = 3.4e38f; bi[t] = 0; }
  for (int j = 0; j < 128; j++) {
    float d2 = 0.f;
    for (int c = 0; c < DC; c++) {
      float df = LF[tid][c] - LF[j][c];
      d2 += df * df;
    }
    float v = d2; int vi = j;
#pragma unroll
    for (int t = 0; t < 16; t++) {
      if (v < bd[t]) {
        float tv = bd[t]; int ti = bi[t];
        bd[t] = v; bi[t] = vi; v = tv; vi = ti;
      }
    }
  }
#pragma unroll
  for (int t = 0; t < 16; t++)
    IDX[((size_t)g * 128 + tid) * 16 + t] = bi[t];
}

// =====================================================================
// Pack fp32 weights [din x dh] into WMMA B-fragment layout, f16.
// Tile (kt,nt) = 32x16; lane L holds col nt*16+(L&15); 16 contiguous
// halves per lane: k_local = ((L>>4)<<3) + e + (e>=8 ? 8 : 0).
// =====================================================================
__global__ void pack_w(const float* __restrict__ W, _Float16* __restrict__ out,
                       int din, int dh) {
  int id = blockIdx.x * blockDim.x + threadIdx.x;
  int e = id & 15, lane = (id >> 4) & 31, tile = id >> 9;
  int NT = dh / 16;
  int nt = tile % NT, kt = tile / NT;
  int kb = (lane >> 4) << 3;
  int kl = kb + e + ((e >= 8) ? 8 : 0);
  int k = kt * 32 + kl;
  int n = nt * 16 + (lane & 15);
  float v = (k < din) ? W[k * dh + n] : 0.f;
  out[id] = (_Float16)v;
}

// =====================================================================
// WMMA helpers
// =====================================================================
template <int NT>
__device__ __forceinline__ void zero_acc(v8f* acc) {
#pragma unroll
  for (int nt = 0; nt < NT; nt++)
#pragma unroll
    for (int e = 0; e < 8; e++) acc[nt][e] = 0.f;
}

// D = A(16 x KT*32, LDS, row stride cstride) x B(packed), accumulate
template <int KT, int NT>
__device__ __forceinline__ void gemm_tile(const _Float16* Abase, int cstride,
                                          const _Float16* __restrict__ Bp,
                                          int lane, v8f* acc) {
  int m = lane & 15;
  int kb = (lane >> 4) << 3;
  const _Float16* arow = Abase + m * cstride;
#pragma unroll
  for (int kt = 0; kt < KT; kt++) {
    const _Float16* ab = arow + kt * 32 + kb;
    v16h av;
#pragma unroll
    for (int e = 0; e < 8; e++) { av[e] = ab[e]; av[8 + e] = ab[e + 16]; }
#pragma unroll
    for (int nt = 0; nt < NT; nt++) {
      v16h bv = ((const v16h*)Bp)[(kt * NT + nt) * 32 + lane];
      acc[nt] = __builtin_amdgcn_wmma_f32_16x16x32_f16(
          false, av, false, bv, (short)0, acc[nt], false, false);
    }
  }
}

template <int NT>
__device__ __forceinline__ void stats_accum(const v8f* acc, float* bs1, float* bs2,
                                            int lane) {
#pragma unroll
  for (int nt = 0; nt < NT; nt++) {
    float s = 0.f, q = 0.f;
#pragma unroll
    for (int r = 0; r < 8; r++) { float v = acc[nt][r]; s += v; q += v * v; }
    s += __shfl_xor(s, 16);
    q += __shfl_xor(q, 16);
    if (lane < 16) {
      atomicAdd(&bs1[nt * 16 + lane], s);
      atomicAdd(&bs2[nt * 16 + lane], q);
    }
  }
}

// y = relu(a*acc + sh) -> LDS f16 [16 x cstride]
template <int NT>
__device__ __forceinline__ void bn_relu_store(const v8f* acc,
                                              const float* __restrict__ a,
                                              const float* __restrict__ sh,
                                              _Float16* Hbase, int cstride, int lane) {
  int col0 = lane & 15;
  int rbase = (lane >> 4) << 3;
#pragma unroll
  for (int nt = 0; nt < NT; nt++) {
    int c = nt * 16 + col0;
    float av = a[c], sv = sh[c];
#pragma unroll
    for (int r = 0; r < 8; r++)
      Hbase[(rbase + r) * cstride + c] =
          (_Float16)fmaxf(0.f, av * acc[nt][r] + sv);
  }
}

// gather edge features into LDS A tile [16 x KPAD] f16
template <int KPAD>
__device__ __forceinline__ void build_edges(const float* __restrict__ Fi,
                                            const float* __restrict__ Fg, int DC,
                                            const int* __restrict__ idxn,
                                            _Float16* Abase, int lane) {
  for (int k = 0; k < 16; k++) {
    const float* Fj = Fg + idxn[k] * DC;
    for (int c = lane; c < KPAD; c += 32) {
      float v = (c < DC) ? Fi[c]
                         : ((c < 2 * DC) ? (Fj[c - DC] - Fi[c - DC]) : 0.f);
      Abase[k * KPAD + c] = (_Float16)v;
    }
  }
}

// =====================================================================
// Conv passes: one wave per node (16 edges == one WMMA M-tile)
// =====================================================================
template <int KPAD, int DH, int WAVES>
__global__ __launch_bounds__(WAVES * 32)
void conv_pass1(const float* __restrict__ Fin, int DC, const int* __restrict__ IDX,
                const _Float16* __restrict__ W1p,
                float* __restrict__ S1, float* __restrict__ S2) {
  constexpr int KT = KPAD / 32, NT = DH / 16;
  __shared__ __align__(16) _Float16 A[WAVES][16][KPAD];
  __shared__ float bs1[DH], bs2[DH];
  const int tid = threadIdx.x, lane = tid & 31, wid = tid >> 5;
  for (int c = tid; c < DH; c += WAVES * 32) { bs1[c] = 0.f; bs2[c] = 0.f; }
  __syncthreads();
  const int node = blockIdx.x * WAVES + wid;
  const int g = node >> 7;
  build_edges<KPAD>(Fin + (size_t)node * DC, Fin + (size_t)(g << 7) * DC, DC,
                    IDX + (size_t)node * 16, &A[wid][0][0], lane);
  v8f acc[NT];
  zero_acc<NT>(acc);
  gemm_tile<KT, NT>(&A[wid][0][0], KPAD, W1p, lane, acc);
  stats_accum<NT>(acc, bs1, bs2, lane);
  __syncthreads();
  for (int c = tid; c < DH; c += WAVES * 32) {
    atomicAdd(&S1[c], bs1[c]);
    atomicAdd(&S2[c], bs2[c]);
  }
}

template <int KPAD, int DH, int WAVES>
__global__ __launch_bounds__(WAVES * 32)
void conv_pass2(const float* __restrict__ Fin, int DC, const int* __restrict__ IDX,
                const _Float16* __restrict__ W1p, const _Float16* __restrict__ W2p,
                const float* __restrict__ a1, const float* __restrict__ sh1,
                float* __restrict__ S1, float* __restrict__ S2) {
  constexpr int KT = KPAD / 32, NT = DH / 16, KT2 = DH / 32;
  __shared__ __align__(16) _Float16 A[WAVES][16][KPAD];
  __shared__ __align__(16) _Float16 H[WAVES][16][DH];
  __shared__ float bs1[DH], bs2[DH];
  const int tid = threadIdx.x, lane = tid & 31, wid = tid >> 5;
  for (int c = tid; c < DH; c += WAVES * 32) { bs1[c] = 0.f; bs2[c] = 0.f; }
  __syncthreads();
  const int node = blockIdx.x * WAVES + wid;
  const int g = node >> 7;
  build_edges<KPAD>(Fin + (size_t)node * DC, Fin + (size_t)(g << 7) * DC, DC,
                    IDX + (size_t)node * 16, &A[wid][0][0], lane);
  v8f acc[NT];
  zero_acc<NT>(acc);
  gemm_tile<KT, NT>(&A[wid][0][0], KPAD, W1p, lane, acc);
  bn_relu_store<NT>(acc, a1, sh1, &H[wid][0][0], DH, lane);
  zero_acc<NT>(acc);
  gemm_tile<KT2, NT>(&H[wid][0][0], DH, W2p, lane, acc);
  stats_accum<NT>(acc, bs1, bs2, lane);
  __syncthreads();
  for (int c = tid; c < DH; c += WAVES * 32) {
    atomicAdd(&S1[c], bs1[c]);
    atomicAdd(&S2[c], bs2[c]);
  }
}

template <int KPAD, int DH, int WAVES>
__global__ __launch_bounds__(WAVES * 32)
void conv_pass3(const float* __restrict__ Fin, int DC, const int* __restrict__ IDX,
                const _Float16* __restrict__ W1p, const _Float16* __restrict__ W2p,
                const _Float16* __restrict__ W3p,
                const float* __restrict__ a1, const float* __restrict__ sh1,
                const float* __restrict__ a2, const float* __restrict__ sh2,
                const float* __restrict__ b3, const float* __restrict__ X1,
                float* __restrict__ Fout) {
  constexpr int KT = KPAD / 32, NT = DH / 16, KT2 = DH / 32, FS = 4 + DH;
  __shared__ __align__(16) _Float16 A[WAVES][16][KPAD];
  __shared__ __align__(16) _Float16 H1[WAVES][16][DH];
  __shared__ __align__(16) _Float16 H2[WAVES][16][DH];
  const int tid = threadIdx.x, lane = tid & 31, wid = tid >> 5;
  const int node = blockIdx.x * WAVES + wid;
  const int g = node >> 7;
  build_edges<KPAD>(Fin + (size_t)node * DC, Fin + (size_t)(g << 7) * DC, DC,
                    IDX + (size_t)node * 16, &A[wid][0][0], lane);
  v8f acc[NT];
  zero_acc<NT>(acc);
  gemm_tile<KT, NT>(&A[wid][0][0], KPAD, W1p, lane, acc);
  bn_relu_store<NT>(acc, a1, sh1, &H1[wid][0][0], DH, lane);
  zero_acc<NT>(acc);
  gemm_tile<KT2, NT>(&H1[wid][0][0], DH, W2p, lane, acc);
  bn_relu_store<NT>(acc, a2, sh2, &H2[wid][0][0], DH, lane);
  zero_acc<NT>(acc);
  gemm_tile<KT2, NT>(&H2[wid][0][0], DH, W3p, lane, acc);
  // mean over 16 edges (rows) + bias, write concat(x1, h) node row
#pragma unroll
  for (int nt = 0; nt < NT; nt++) {
    float s = 0.f;
#pragma unroll
    for (int r = 0; r < 8; r++) s += acc[nt][r];
    s += __shfl_xor(s, 16);
    if (lane < 16) {
      int c = nt * 16 + lane;
      Fout[(size_t)node * FS + 4 + c] = s * 0.0625f + b3[c];
    }
  }
  if (lane < 4) Fout[(size_t)node * FS + lane] = X1[(size_t)node * 4 + lane];
}

// =====================================================================
// Pool over nodes + concat u1 -> z [512 x 134]
// =====================================================================
__global__ __launch_bounds__(128)
void pool_concat(const float* __restrict__ F, const float* __restrict__ U1,
                 float* __restrict__ Z) {
  int g = blockIdx.x;
  for (int c = threadIdx.x; c < 132; c += 128) {
    float s = 0.f;
    for (int p = 0; p < 128; p++) s += F[((size_t)(g * 128 + p)) * 132 + c];
    Z[(size_t)g * 134 + 2 + c] = s * (1.f / 128.f);
  }
  if (threadIdx.x < 2) Z[(size_t)g * 134 + threadIdx.x] = U1[g * 2 + threadIdx.x];
}

// =====================================================================
// Host orchestration
// =====================================================================
struct MLPPtrs {
  const float *W1, *b1, *g1, *be1, *W2, *b2, *g2, *be2, *W3, *b3;
};

template <int KPAD, int DH, int WV12, int WV3>
static void run_conv_layer(hipStream_t stream, const float* Fin, int DC, int* IDX,
                           const MLPPtrs& p, _Float16* W1P, _Float16* W2P,
                           _Float16* W3P, float* S1, float* S2, float* A1,
                           float* SH1, float* A2, float* SH2, const float* X1,
                           float* Fout) {
  constexpr int NT = DH / 16, KT1 = KPAD / 32, KT2 = DH / 32;
  const float invN = 1.f / 1048576.f;  // G*P*K edges
  knn_kernel<<<512, 128, 0, stream>>>(Fin, DC, IDX);
  pack_w<<<(KT1 * NT * 512) / 256, 256, 0, stream>>>(p.W1, W1P, 2 * DC, DH);
  pack_w<<<(KT2 * NT * 512) / 256, 256, 0, stream>>>(p.W2, W2P, DH, DH);
  pack_w<<<(KT2 * NT * 512) / 256, 256, 0, stream>>>(p.W3, W3P, DH, DH);
  zero_f<<<1, 256, 0, stream>>>(S1, DH);
  zero_f<<<1, 256, 0, stream>>>(S2, DH);
  conv_pass1<KPAD, DH, WV12><<<65536 / WV12, WV12 * 32, 0, stream>>>(
      Fin, DC, IDX, W1P, S1, S2);
  conv_stats_fin<<<1, 128, 0, stream>>>(S1, S2, p.g1, p.be1, A1, SH1, DH, invN);
  zero_f<<<1, 256, 0, stream>>>(S1, DH);
  zero_f<<<1, 256, 0, stream>>>(S2, DH);
  conv_pass2<KPAD, DH, WV12><<<65536 / WV12, WV12 * 32, 0, stream>>>(
      Fin, DC, IDX, W1P, W2P, A1, SH1, S1, S2);
  conv_stats_fin<<<1, 128, 0, stream>>>(S1, S2, p.g2, p.be2, A2, SH2, DH, invN);
  conv_pass3<KPAD, DH, WV3><<<65536 / WV3, WV3 * 32, 0, stream>>>(
      Fin, DC, IDX, W1P, W2P, W3P, A1, SH1, A2, SH2, p.b3, X1, Fout);
}

extern "C" void kernel_launch(void* const* d_in, const int* in_sizes, int n_in,
                              void* d_out, int out_size, void* d_ws, size_t ws_size,
                              hipStream_t stream) {
  (void)in_sizes; (void)n_in; (void)out_size; (void)ws_size;
  const float* X = (const float*)d_in[0];   // [65536,4]
  const float* U = (const float*)d_in[1];   // [512,2]
  const float* bnxg = (const float*)d_in[2];
  const float* bnxb = (const float*)d_in[3];
  const float* bnug = (const float*)d_in[4];
  const float* bnub = (const float*)d_in[5];
  auto mk = [&](int base) {
    MLPPtrs m;
    m.W1 = (const float*)d_in[base + 0]; m.b1 = (const float*)d_in[base + 1];
    m.g1 = (const float*)d_in[base + 2]; m.be1 = (const float*)d_in[base + 3];
    m.W2 = (const float*)d_in[base + 4]; m.b2 = (const float*)d_in[base + 5];
    m.g2 = (const float*)d_in[base + 6]; m.be2 = (const float*)d_in[base + 7];
    m.W3 = (const float*)d_in[base + 8]; m.b3 = (const float*)d_in[base + 9];
    return m;
  };
  MLPPtrs c1 = mk(6), c2 = mk(16), c3 = mk(26), om = mk(36);

  // workspace layout
  uintptr_t base = (uintptr_t)d_ws;
  size_t off = 0;
  auto alloc = [&](size_t bytes) -> void* {
    void* p = (void*)(base + off);
    off = (off + bytes + 255) & ~(size_t)255;
    return p;
  };
  float* X1 = (float*)alloc((size_t)65536 * 4 * 4);
  float* U1 = (float*)alloc(512 * 2 * 4);
  int* IDX = (int*)alloc((size_t)65536 * 16 * 4);
  float* F2 = (float*)alloc((size_t)65536 * 36 * 4);
  float* F3 = (float*)alloc((size_t)65536 * 68 * 4);
  float* F4 = (float*)alloc((size_t)65536 * 132 * 4);
  _Float16* W1P = (_Float16*)alloc(65536);  // 64KB (max 40960B needed)
  _Float16* W2P = (_Float16*)alloc(65536);
  _Float16* W3P = (_Float16*)alloc(65536);
  float* S1 = (float*)alloc(512);
  float* S2 = (float*)alloc(512);
  float* A1 = (float*)alloc(512);
  float* SH1 = (float*)alloc(512);
  float* A2 = (float*)alloc(512);
  float* SH2 = (float*)alloc(512);
  float* Z = (float*)alloc((size_t)512 * 134 * 4);
  float* T1 = (float*)alloc((size_t)512 * 256 * 4);
  float* T2 = (float*)alloc((size_t)512 * 256 * 4);
  float* OA = (float*)alloc(256 * 4);
  float* OSH = (float*)alloc(256 * 4);

  // BN(x) -> X1 ; BN(u) -> U1
  zero_f<<<1, 256, 0, stream>>>(S1, 4);
  zero_f<<<1, 256, 0, stream>>>(S2, 4);
  colstats<<<256, 256, 0, stream>>>(X, 65536, 4, S1, S2);
  colnorm<<<(65536 * 4 + 255) / 256, 256, 0, stream>>>(X, 65536, 4, S1, S2,
                                                       bnxg, bnxb, X1);
  zero_f<<<1, 256, 0, stream>>>(S1, 2);
  zero_f<<<1, 256, 0, stream>>>(S2, 2);
  colstats<<<4, 256, 0, stream>>>(U, 512, 2, S1, S2);
  colnorm<<<4, 256, 0, stream>>>(U, 512, 2, S1, S2, bnug, bnub, U1);

  // three dynamic EdgeConv layers (WMMA)
  run_conv_layer<32, 32, 4, 4>(stream, X, 4, IDX, c1, W1P, W2P, W3P, S1, S2, A1,
                               SH1, A2, SH2, X1, F2);
  run_conv_layer<96, 64, 4, 4>(stream, F2, 36, IDX, c2, W1P, W2P, W3P, S1, S2,
                               A1, SH1, A2, SH2, X1, F3);
  run_conv_layer<160, 128, 4, 2>(stream, F3, 68, IDX, c3, W1P, W2P, W3P, S1, S2,
                                 A1, SH1, A2, SH2, X1, F4);

  // graph pooling + output MLP (fp32, tiny)
  pool_concat<<<512, 128, 0, stream>>>(F4, U1, Z);
  gemm_bias<<<(512 * 256 + 255) / 256, 256, 0, stream>>>(Z, om.W1, om.b1, T1,
                                                         512, 134, 256);
  bn_stats_rows<<<256, 256, 0, stream>>>(T1, 512, 256, om.g1, om.be1, OA, OSH);
  bn_relu<<<512, 256, 0, stream>>>(T1, 512 * 256, 256, OA, OSH);
  gemm_bias<<<(512 * 256 + 255) / 256, 256, 0, stream>>>(T1, om.W2, om.b2, T2,
                                                         512, 256, 256);
  bn_stats_rows<<<256, 256, 0, stream>>>(T2, 512, 256, om.g2, om.be2, OA, OSH);
  bn_relu<<<512, 256, 0, stream>>>(T2, 512 * 256, 256, OA, OSH);
  final_out<<<512, 256, 0, stream>>>(T2, om.W3, om.b3, (float*)d_out);
}
